// MultiHeadAttention_79242146611868
// MI455X (gfx1250) — compile-verified, matrix operand order checked by reading, not run on hardware
//
#include <hip/hip_runtime.h>

typedef __bf16 bf16;
typedef __bf16 bf16x16 __attribute__((ext_vector_type(16)));
typedef float  f32x8   __attribute__((ext_vector_type(8)));
typedef int    v4i     __attribute__((vector_size(16)));
typedef __attribute__((address_space(1))) v4i* gv4i_p;   // global int4*
typedef __attribute__((address_space(3))) v4i* lv4i_p;   // LDS int4*

#define D_MODEL 1024
#define NHEAD   16
#define DK      64

#if __has_builtin(__builtin_amdgcn_global_load_async_to_lds_b128)
#define HAS_ASYNC_LDS 1
#else
#define HAS_ASYNC_LDS 0
#endif

// ---------- helpers ----------
__device__ __forceinline__ bf16 f2bf(float f) {
    union { float f; unsigned u; } a; a.f = f;
    unsigned u = a.u;
    unsigned r = (u + 0x7FFFu + ((u >> 16) & 1u)) >> 16;
    union { unsigned short s; bf16 b; } o; o.s = (unsigned short)r;
    return o.b;
}

#if HAS_ASYNC_LDS
__device__ __forceinline__ void copy16_async(const bf16* g, bf16* l) {
    __builtin_amdgcn_global_load_async_to_lds_b128(
        (gv4i_p)(void*)g, (lv4i_p)(void*)l, 0, 0);
}
__device__ __forceinline__ void wait_async0() {
#if __has_builtin(__builtin_amdgcn_s_wait_asynccnt)
    __builtin_amdgcn_s_wait_asynccnt(0);
#else
    asm volatile("s_wait_asynccnt 0x0" ::: "memory");
#endif
}
#define COPY16(g, l) copy16_async((g), (l))
#define WAIT_STAGE() wait_async0()
#else
__device__ __forceinline__ void copy16_sync(const bf16* g, bf16* l) {
    *(uint4*)l = *(const uint4*)g;
}
#define COPY16(g, l) copy16_sync((g), (l))
#define WAIT_STAGE() ((void)0)
#endif

// A/B fragment loader for v_wmma_f32_16x16x32_bf16.
// Per ISA 7.12.2: lane half `hi`, VGPR j holds K pair at
// kb = (j<4 ? 0 : 16) + hi*8 + (j&3)*2. `base` is a row-major tile whose
// second index is the contraction (K) dimension.
__device__ __forceinline__ bf16x16 load_frag(const bf16* base, int row, int ld,
                                             int kk, int hi) {
    union { bf16x16 v; unsigned u[8]; } f;
#pragma unroll
    for (int j = 0; j < 8; ++j) {
        int kb = ((j < 4) ? 0 : 16) + hi * 8 + (j & 3) * 2;
        f.u[j] = *(const unsigned*)(base + row * ld + kk + kb);
    }
    return f.v;
}

__device__ __forceinline__ f32x8 wmma_bf16(bf16x16 a, bf16x16 b, f32x8 c) {
    return __builtin_amdgcn_wmma_f32_16x16x32_bf16(
        /*neg_a=*/false, a, /*neg_b=*/false, b,
        /*c_mod=*/(short)0, c, /*reuse_a=*/false, /*reuse_b=*/false);
}

__device__ __forceinline__ float redmax16(float t) {
#pragma unroll
    for (int o = 1; o < 16; o <<= 1) t = fmaxf(t, __shfl_xor(t, o, 32));
    return t;
}
__device__ __forceinline__ float redsum16(float t) {
#pragma unroll
    for (int o = 1; o < 16; o <<= 1) t += __shfl_xor(t, o, 32);
    return t;
}

// ---------- elementwise conversion ----------
__global__ void cvt_f32_bf16(const float* __restrict__ src,
                             bf16* __restrict__ dst, int n) {
    for (int i = blockIdx.x * blockDim.x + threadIdx.x; i < n;
         i += gridDim.x * blockDim.x)
        dst[i] = f2bf(src[i]);
}

// dst[n][k] = src[k][n]  (square dim x dim), bf16 out
__global__ void transpose_bf16(const float* __restrict__ src,
                               bf16* __restrict__ dst, int dim) {
    int idx = blockIdx.x * blockDim.x + threadIdx.x;
    if (idx < dim * dim) {
        int k = idx % dim;
        int n = idx / dim;
        dst[(size_t)n * dim + k] = f2bf(src[(size_t)k * dim + n]);
    }
}

// ---------- tiled WMMA GEMM:  out = A[M,K] * Wt[N,K]^T + bias ----------
// Double-buffered LDS staging (async global->LDS when available).
// mode 0: store bf16 head-split   [(b*H+h)*S + s]*64 + dk
// mode 1: store bf16 head-split V^T [(b*H+h)*64 + dk]*S + s
// mode 2: store f32 row-major [m*N + n]
__global__ __launch_bounds__(256) void gemm_bf16(
    const bf16* __restrict__ A, const bf16* __restrict__ Wt,
    const float* __restrict__ bias, void* __restrict__ out,
    int M, int N, int K, int S, int H, int mode) {
    __shared__ bf16 Alds[2][128 * 64];
    __shared__ bf16 Blds[2][128 * 64];

    const int tid  = threadIdx.x;
    const int lane = tid & 31;
    const int wave = tid >> 5;        // 8 waves
    const int hi   = lane >> 4;
    const int r    = lane & 15;
    const int wm   = wave >> 1;       // 0..3  (M direction, 32 rows each)
    const int wn   = wave & 1;        // 0..1  (N direction, 64 cols each)
    const int m0   = blockIdx.y * 128;
    const int n0   = blockIdx.x * 128;

    f32x8 acc[2][4];
#pragma unroll
    for (int ms = 0; ms < 2; ++ms)
#pragma unroll
        for (int ns = 0; ns < 4; ++ns)
#pragma unroll
            for (int e = 0; e < 8; ++e) acc[ms][ns][e] = 0.0f;

    const int T = K / 64;
    // stage tile t into buffer (t & 1): 128x64 bf16 of A and of Wt
#define GEMM_STAGE(t)                                                         \
    do {                                                                      \
        int _b = (t) & 1;                                                     \
        _Pragma("unroll") for (int i = 0; i < 4; ++i) {                       \
            int c = tid + i * 256;                                            \
            int row = c >> 3, col8 = (c & 7) * 8;                             \
            COPY16(&A[(size_t)(m0 + row) * K + (t) * 64 + col8],              \
                   &Alds[_b][row * 64 + col8]);                               \
            COPY16(&Wt[(size_t)(n0 + row) * K + (t) * 64 + col8],             \
                   &Blds[_b][row * 64 + col8]);                               \
        }                                                                     \
    } while (0)

    GEMM_STAGE(0);
    for (int t = 0; t < T; ++t) {
        WAIT_STAGE();        // tile t resident in LDS (per-wave ASYNCcnt)
        __syncthreads();     // all waves' fills visible / prior reads done
        if (t + 1 < T) GEMM_STAGE(t + 1);   // overlap with compute
        const bf16* Ab = Alds[t & 1];
        const bf16* Bb = Blds[t & 1];
#pragma unroll
        for (int kk = 0; kk < 64; kk += 32) {
            bf16x16 a[2];
#pragma unroll
            for (int ms = 0; ms < 2; ++ms)
                a[ms] = load_frag(Ab, wm * 32 + ms * 16 + r, 64, kk, hi);
#pragma unroll
            for (int ns = 0; ns < 4; ++ns) {
                bf16x16 b = load_frag(Bb, wn * 64 + ns * 16 + r, 64, kk, hi);
#pragma unroll
                for (int ms = 0; ms < 2; ++ms)
                    acc[ms][ns] = wmma_bf16(a[ms], b, acc[ms][ns]);
            }
        }
    }
#undef GEMM_STAGE

    // epilogue: C fragment VGPR v -> row v+8*hi, lane -> column
#pragma unroll
    for (int ms = 0; ms < 2; ++ms)
#pragma unroll
        for (int ns = 0; ns < 4; ++ns)
#pragma unroll
            for (int v = 0; v < 8; ++v) {
                int m = m0 + wm * 32 + ms * 16 + v + 8 * hi;
                int n = n0 + wn * 64 + ns * 16 + r;
                float val = acc[ms][ns][v] + bias[n];
                if (mode == 2) {
                    ((float*)out)[(size_t)m * N + n] = val;
                } else {
                    int b  = m / S, s = m % S;
                    int h  = n >> 6, dk = n & 63;
                    if (mode == 0)
                        ((bf16*)out)[(((size_t)(b * H + h)) * S + s) * 64 + dk] =
                            f2bf(val);
                    else
                        ((bf16*)out)[(((size_t)(b * H + h)) * 64 + dk) * S + s] =
                            f2bf(val);
                }
            }
}

// ---------- flash attention (online softmax, WMMA for QK^T and P*V) ------
// Qp,Kp: [B*H, S, 64] bf16; Vt: [B*H, 64, S] bf16; mask: [B,1,S,S] int32
// attnout: [B*S, H*64] bf16
__global__ __launch_bounds__(128) void attn_fused(
    const bf16* __restrict__ Qp, const bf16* __restrict__ Kp,
    const bf16* __restrict__ Vt, const int* __restrict__ mask,
    bf16* __restrict__ attnout, int S, int H) {
    __shared__ bf16 Qs[64 * 64];
    __shared__ bf16 Ks[2][64 * 64];       // Ks[b][key][d]
    __shared__ bf16 Vs[2][64 * 64];       // Vs[b][dd][key]
    __shared__ bf16 Ps[4 * 16 * 64];      // per-wave P tile [16][64]

    const int tid  = threadIdx.x;
    const int lane = tid & 31;
    const int wave = tid >> 5;            // 4 waves, 16 query rows each
    const int hi   = lane >> 4;
    const int r    = lane & 15;
    const int qt   = blockIdx.x;          // query tile (64 rows)
    const int bh   = blockIdx.y;          // batch*head
    const int bidx = bh / H;
    const int h    = bh % H;

    const bf16* Qh = Qp + (size_t)bh * S * 64;
    const bf16* Kh = Kp + (size_t)bh * S * 64;
    const bf16* Vh = Vt + (size_t)bh * 64 * S;

    // stage Q tile (64x64) -- visibility covered by first loop barrier
#pragma unroll
    for (int i = 0; i < 4; ++i) {
        int c = tid + i * 128;
        int row = c >> 3, col8 = (c & 7) * 8;
        *(uint4*)&Qs[row * 64 + col8] =
            *(const uint4*)&Qh[((size_t)(qt * 64 + row)) * 64 + col8];
    }

    f32x8 o[4];
#pragma unroll
    for (int ns = 0; ns < 4; ++ns)
#pragma unroll
        for (int e = 0; e < 8; ++e) o[ns][e] = 0.0f;
    float mrow[8], lrow[8];
#pragma unroll
    for (int v = 0; v < 8; ++v) { mrow[v] = -1e30f; lrow[v] = 0.0f; }

    bf16* Pw = Ps + wave * 16 * 64;
    const int nkt = S / 64;

#define ATTN_STAGE(kt)                                                        \
    do {                                                                      \
        int _b = (kt) & 1;                                                    \
        _Pragma("unroll") for (int i = 0; i < 4; ++i) {                       \
            int c = tid + i * 128;                                            \
            int row = c >> 3, col8 = (c & 7) * 8;                             \
            COPY16(&Kh[((size_t)((kt) * 64 + row)) * 64 + col8],              \
                   &Ks[_b][row * 64 + col8]);                                 \
            COPY16(&Vh[(size_t)row * S + (kt) * 64 + col8],                   \
                   &Vs[_b][row * 64 + col8]);                                 \
        }                                                                     \
    } while (0)

    ATTN_STAGE(0);
    for (int kt = 0; kt < nkt; ++kt) {
        WAIT_STAGE();
        __syncthreads();
        if (kt + 1 < nkt) ATTN_STAGE(kt + 1);
        const bf16* Kb = Ks[kt & 1];
        const bf16* Vb = Vs[kt & 1];

        // scores: 16 (q) x 64 (keys), contraction over d=64
        f32x8 sc[4];
#pragma unroll
        for (int ns = 0; ns < 4; ++ns)
#pragma unroll
            for (int e = 0; e < 8; ++e) sc[ns][e] = 0.0f;
#pragma unroll
        for (int kk = 0; kk < 64; kk += 32) {
            bf16x16 a = load_frag(Qs, wave * 16 + r, 64, kk, hi);
#pragma unroll
            for (int ns = 0; ns < 4; ++ns) {
                bf16x16 b = load_frag(Kb, ns * 16 + r, 64, kk, hi);
                sc[ns] = wmma_bf16(a, b, sc[ns]);
            }
        }

        // scale + mask
        const float scale = 0.125f;   // 1/sqrt(64)
#pragma unroll
        for (int ns = 0; ns < 4; ++ns)
#pragma unroll
            for (int v = 0; v < 8; ++v) {
                int qg = qt * 64 + wave * 16 + v + 8 * hi;
                int kg = kt * 64 + ns * 16 + r;
                int mv = mask[((size_t)bidx * S + qg) * S + kg];
                sc[ns][v] = (mv == 0) ? -1e30f : sc[ns][v] * scale;
            }

        // online softmax update per row (row v+8*hi; columns across 16 lanes)
#pragma unroll
        for (int v = 0; v < 8; ++v) {
            float t = fmaxf(fmaxf(sc[0][v], sc[1][v]),
                            fmaxf(sc[2][v], sc[3][v]));
            t = redmax16(t);
            float mn = fmaxf(mrow[v], t);
            float alpha = __expf(mrow[v] - mn);
            float ssum = 0.0f;
#pragma unroll
            for (int ns = 0; ns < 4; ++ns) {
                float p = __expf(sc[ns][v] - mn);
                sc[ns][v] = p;
                ssum += p;
            }
            ssum = redsum16(ssum);
            lrow[v] = lrow[v] * alpha + ssum;
            mrow[v] = mn;
#pragma unroll
            for (int ns = 0; ns < 4; ++ns) o[ns][v] *= alpha;
        }

        // spill P (C layout) to LDS so it can be refetched in A layout
#pragma unroll
        for (int ns = 0; ns < 4; ++ns)
#pragma unroll
            for (int v = 0; v < 8; ++v)
                Pw[(v + 8 * hi) * 64 + ns * 16 + r] = f2bf(sc[ns][v]);
        __syncthreads();

        // O += P[16,64] * V[64,64]
#pragma unroll
        for (int kk = 0; kk < 64; kk += 32) {
            bf16x16 a = load_frag(Pw, r, 64, kk, hi);
#pragma unroll
            for (int ns = 0; ns < 4; ++ns) {
                bf16x16 b = load_frag(Vb, ns * 16 + r, 64, kk, hi);
                o[ns] = wmma_bf16(a, b, o[ns]);
            }
        }
    }
#undef ATTN_STAGE

    // normalize and store in [b, s, h*64 + dd] layout (bf16)
#pragma unroll
    for (int ns = 0; ns < 4; ++ns)
#pragma unroll
        for (int v = 0; v < 8; ++v) {
            int qg = qt * 64 + wave * 16 + v + 8 * hi;
            int dd = ns * 16 + r;
            float val = o[ns][v] / lrow[v];
            attnout[((size_t)(bidx * S + qg)) * (H * 64) + h * 64 + dd] =
                f2bf(val);
        }
}

// ---------- launch ----------
extern "C" void kernel_launch(void* const* d_in, const int* in_sizes, int n_in,
                              void* d_out, int out_size, void* d_ws,
                              size_t ws_size, hipStream_t stream) {
    const float* q  = (const float*)d_in[0];
    const float* k  = (const float*)d_in[1];
    const float* v  = (const float*)d_in[2];
    const int* mask = (const int*)d_in[3];
    const float* wq = (const float*)d_in[4];
    const float* bq = (const float*)d_in[5];
    const float* wk = (const float*)d_in[6];
    const float* bk = (const float*)d_in[7];
    const float* wv = (const float*)d_in[8];
    const float* bv = (const float*)d_in[9];
    const float* wo = (const float*)d_in[10];
    const float* bo = (const float*)d_in[11];

    const int D  = D_MODEL;
    const int H  = NHEAD;
    const int MS = in_sizes[0] / D;      // B*S
    const int S  = in_sizes[3] / MS;     // mask is B*S*S

    char* p = (char*)d_ws;
    const size_t szA = (size_t)MS * D * sizeof(bf16);   // bf16 activation
    const size_t szW = (size_t)D * D * sizeof(bf16);    // bf16 weight
    bf16* qb   = (bf16*)p; p += szA;
    bf16* kb   = (bf16*)p; p += szA;
    bf16* vb   = (bf16*)p; p += szA;
    bf16* wqT  = (bf16*)p; p += szW;
    bf16* wkT  = (bf16*)p; p += szW;
    bf16* wvT  = (bf16*)p; p += szW;
    bf16* woT  = (bf16*)p; p += szW;
    bf16* Qp   = (bf16*)p; p += szA;    // [B*H, S, 64]
    bf16* Kp   = (bf16*)p; p += szA;    // [B*H, S, 64]
    bf16* Vt   = (bf16*)p; p += szA;    // [B*H, 64, S]
    bf16* aout = (bf16*)p; p += szA;    // [B*S, H*64]

    // 1) precision conversion / weight transpose
    const int nAct = MS * D;
    cvt_f32_bf16<<<2048, 256, 0, stream>>>(q, qb, nAct);
    cvt_f32_bf16<<<2048, 256, 0, stream>>>(k, kb, nAct);
    cvt_f32_bf16<<<2048, 256, 0, stream>>>(v, vb, nAct);
    const int nW = D * D;
    transpose_bf16<<<(nW + 255) / 256, 256, 0, stream>>>(wq, wqT, D);
    transpose_bf16<<<(nW + 255) / 256, 256, 0, stream>>>(wk, wkT, D);
    transpose_bf16<<<(nW + 255) / 256, 256, 0, stream>>>(wv, wvT, D);
    transpose_bf16<<<(nW + 255) / 256, 256, 0, stream>>>(wo, woT, D);

    // 2) input projections (WMMA GEMMs)
    dim3 gg(D / 128, MS / 128);
    gemm_bf16<<<gg, 256, 0, stream>>>(qb, wqT, bq, Qp, MS, D, D, S, H, 0);
    gemm_bf16<<<gg, 256, 0, stream>>>(kb, wkT, bk, Kp, MS, D, D, S, H, 0);
    gemm_bf16<<<gg, 256, 0, stream>>>(vb, wvT, bv, Vt, MS, D, D, S, H, 1);

    // 3) fused masked flash attention
    dim3 ga(S / 64, (MS / S) * H);
    attn_fused<<<ga, 128, 0, stream>>>(Qp, Kp, Vt, mask, aout, S, H);

    // 4) output projection -> fp32 d_out
    gemm_bf16<<<gg, 256, 0, stream>>>(aout, woT, bo, (float*)d_out,
                                      MS, D, D, S, H, 2);
}
